// MaskedCausalAttention_55336358642962
// MI455X (gfx1250) — matrix operand order repaired and checked by reading, not compiled, not run on hardware
//
#include <hip/hip_runtime.h>
#include <hip/hip_bf16.h>

// MI455X / gfx1250 causal multi-head attention, f16 WMMA path, f32 accumulate.
// B=2, T=2048, C=1024, N=16 heads, D=64. Needs ~40 MiB workspace.

typedef __attribute__((ext_vector_type(16))) _Float16 v16h;
typedef __attribute__((ext_vector_type(8)))  _Float16 v8h;
typedef __attribute__((ext_vector_type(8)))  float    v8f;

#define BATCH 2
#define SEQ   2048
#define CH    1024
#define NH    16
#define HD    64
#define MROWS (BATCH * SEQ)   // 4096

__device__ inline v8f wmma_f16(v16h a, v16h b, v8f c) {
  return __builtin_amdgcn_wmma_f32_16x16x32_f16(false, a, false, b, (short)0, c,
                                                false, false);
}

// A fragment: 16x32 f16, row-major source with leading dim `ld`.
// Lanes 0-15: row = lane, K = {0..7, 16..23}; lanes 16-31: row = lane-16,
// K = {8..15, 24..31}  (per ISA 16-bit A-matrix table).
__device__ inline v16h load_a_frag(const _Float16* __restrict__ p, int row0,
                                   int k0, int ld, int lane) {
  const int r  = row0 + (lane & 15);
  const int kb = k0 + ((lane & 16) ? 8 : 0);
  const _Float16* base = p + (size_t)r * ld + kb;
  v8h lo = *(const v8h*)(base);
  v8h hi = *(const v8h*)(base + 16);
  v16h f;
#pragma unroll
  for (int j = 0; j < 8; ++j) { f[j] = lo[j]; f[8 + j] = hi[j]; }
  return f;
}

// B fragment: 32x16 f16, loaded from K-contiguous (transposed) storage Bt[n][k]
// with leading dim `ld`. Lanes 0-15 hold K={0..7, 16..23}, lanes 16-31 hold
// K={8..15, 24..31}, column n = lane&15 (matches the A-fragment K order).
__device__ inline v16h load_b_frag(const _Float16* __restrict__ p, int n0,
                                   int k0, int ld, int lane) {
  const int n  = n0 + (lane & 15);
  const int kb = k0 + ((lane & 16) ? 8 : 0);
  const _Float16* base = p + (size_t)n * ld + kb;
  v8h lo = *(const v8h*)(base);
  v8h hi = *(const v8h*)(base + 16);
  v16h f;
#pragma unroll
  for (int j = 0; j < 8; ++j) { f[j] = lo[j]; f[8 + j] = hi[j]; }
  return f;
}

// Reductions across a 16-lane half (xor masks < 16 never cross halves).
__device__ inline float half16_max(float v) {
  v = fmaxf(v, __shfl_xor(v, 1, 32));
  v = fmaxf(v, __shfl_xor(v, 2, 32));
  v = fmaxf(v, __shfl_xor(v, 4, 32));
  v = fmaxf(v, __shfl_xor(v, 8, 32));
  return v;
}
__device__ inline float half16_sum(float v) {
  v += __shfl_xor(v, 1, 32);
  v += __shfl_xor(v, 2, 32);
  v += __shfl_xor(v, 4, 32);
  v += __shfl_xor(v, 8, 32);
  return v;
}

// ---------------- conversion kernels ----------------
__global__ void cvt_f16_kernel(const float* __restrict__ in,
                               _Float16* __restrict__ out, int n) {
  for (int i = blockIdx.x * blockDim.x + threadIdx.x; i < n;
       i += gridDim.x * blockDim.x)
    out[i] = (_Float16)in[i];
}

// W is [K=CH][N=CH] row-major; emit Wt[n][k] f16 so B-fragments are contiguous.
__global__ void cvt_wT_kernel(const float* __restrict__ in,
                              _Float16* __restrict__ out, int n) {
  for (int i = blockIdx.x * blockDim.x + threadIdx.x; i < n;
       i += gridDim.x * blockDim.x) {
    int kk = i >> 10;          // row in W
    int nn = i & (CH - 1);     // col in W
    out[(size_t)nn * CH + kk] = (_Float16)in[i];
  }
}

// ---------------- QKV projection GEMM (32x32 tile / wave) ----------------
// A: [4096 x 1024] f16, Bt: [1024 x 1024] f16 (N-major), bias f32[1024].
// vmode 0: dst = [B,N,T,D] f16.  vmode 1: dst = [B,N,D,T] f16 (V transposed).
// 2x2 register blocking: 4 WMMAs per k-step on 2 A-frags + 2 B-frags.
__global__ __launch_bounds__(256) void gemm_qkv_kernel(
    const _Float16* __restrict__ A, const _Float16* __restrict__ Bt,
    const float* __restrict__ bias, _Float16* __restrict__ dst, int vmode) {
  const int lane = threadIdx.x & 31;
  const int gw   = blockIdx.x * 8 + (threadIdx.x >> 5);   // 32x32 tile id
  const int m0   = (gw >> 5) * 32;                        // 128 M-tiles
  const int n0   = (gw & 31) * 32;                        // 32 N-tiles
  v8f c00 = {}, c01 = {}, c10 = {}, c11 = {};
  for (int k0 = 0; k0 < CH; k0 += 32) {
    // pull the k-block 4 iterations ahead into cache (global_prefetch_b8)
    __builtin_prefetch(A  + (size_t)(m0 + lane) * CH + k0 + 128, 0, 1);
    __builtin_prefetch(Bt + (size_t)(n0 + lane) * CH + k0 + 128, 0, 1);
    v16h a0 = load_a_frag(A, m0,      k0, CH, lane);
    v16h a1 = load_a_frag(A, m0 + 16, k0, CH, lane);
    v16h b0 = load_b_frag(Bt, n0,      k0, CH, lane);
    v16h b1 = load_b_frag(Bt, n0 + 16, k0, CH, lane);
    c00 = wmma_f16(a0, b0, c00);
    c01 = wmma_f16(a0, b1, c01);
    c10 = wmma_f16(a1, b0, c10);
    c11 = wmma_f16(a1, b1, c11);
  }
  const int rowoff = (lane & 16) ? 8 : 0;
  const int col0 = n0 + (lane & 15), col1 = col0 + 16;
  const float bv0 = bias[col0], bv1 = bias[col1];
  const int h0 = col0 >> 6, d0 = col0 & (HD - 1);
  const int h1 = col1 >> 6, d1 = col1 & (HD - 1);
#pragma unroll
  for (int r = 0; r < 8; ++r) {
    const int mr0 = m0 + r + rowoff;      // tile row block 0
    const int mr1 = mr0 + 16;             // tile row block 1
    const int b0i = mr0 >> 11, t0 = mr0 & (SEQ - 1);
    const int b1i = mr1 >> 11, t1 = mr1 & (SEQ - 1);
    const float v00 = c00[r] + bv0, v01 = c01[r] + bv1;
    const float v10 = c10[r] + bv0, v11 = c11[r] + bv1;
    if (vmode == 0) {
      dst[((size_t)(b0i * NH + h0) * SEQ + t0) * HD + d0] = (_Float16)v00;
      dst[((size_t)(b0i * NH + h1) * SEQ + t0) * HD + d1] = (_Float16)v01;
      dst[((size_t)(b1i * NH + h0) * SEQ + t1) * HD + d0] = (_Float16)v10;
      dst[((size_t)(b1i * NH + h1) * SEQ + t1) * HD + d1] = (_Float16)v11;
    } else {
      dst[((size_t)(b0i * NH + h0) * HD + d0) * SEQ + t0] = (_Float16)v00;
      dst[((size_t)(b0i * NH + h1) * HD + d1) * SEQ + t0] = (_Float16)v01;
      dst[((size_t)(b1i * NH + h0) * HD + d0) * SEQ + t1] = (_Float16)v10;
      dst[((size_t)(b1i * NH + h1) * HD + d1) * SEQ + t1] = (_Float16)v11;
    }
  }
}

// ---------------- flash attention ----------------
// q,k: [B,N,T,D] f16; vt: [B,N,D,T] f16; aout: [B,T,C] f16.
__global__ __launch_bounds__(256) void attn_kernel(
    const _Float16* __restrict__ q, const _Float16* __restrict__ k,
    const _Float16* __restrict__ vt, _Float16* __restrict__ aout) {
  __shared__ __align__(16) _Float16 pbuf[8][16 * 32];  // per-wave P staging

  const int lane = threadIdx.x & 31;
  const int wid  = threadIdx.x >> 5;
  const int gw   = blockIdx.x * 8 + wid;     // [0, 4096)
  const int qt   = gw & 127;                 // query tile
  const int hn   = (gw >> 7) & (NH - 1);
  const int bb   = gw >> 11;

  const size_t hoff = (size_t)(bb * NH + hn) * SEQ * HD;
  const _Float16* qh = q  + hoff;
  const _Float16* kh = k  + hoff;
  const _Float16* vh = vt + hoff;            // [HD][SEQ]

  const v16h qa0 = load_a_frag(qh, qt * 16, 0,  HD, lane);
  const v16h qa1 = load_a_frag(qh, qt * 16, 32, HD, lane);

  v8f acc0 = {}, acc1 = {}, acc2 = {}, acc3 = {};
  float m[8], l[8];
#pragma unroll
  for (int r = 0; r < 8; ++r) { m[r] = -__builtin_inff(); l[r] = 0.f; }

  const int rowoff = (lane & 16) ? 8 : 0;
  const int col    = lane & 15;
  const float scale = 0.125f;                // 1/sqrt(64)
  const int tmax = qt * 16 + 15;

  for (int s0 = 0; s0 <= tmax; s0 += 32) {   // wave-uniform loop
    // scores for 16 queries x 32 keys (keys contiguous in K's D-dim)
    v8f slo = {}, shi = {};
    slo = wmma_f16(qa0, load_b_frag(kh, s0,      0,  HD, lane), slo);
    slo = wmma_f16(qa1, load_b_frag(kh, s0,      32, HD, lane), slo);
    shi = wmma_f16(qa0, load_b_frag(kh, s0 + 16, 0,  HD, lane), shi);
    shi = wmma_f16(qa1, load_b_frag(kh, s0 + 16, 32, HD, lane), shi);

#pragma unroll
    for (int r = 0; r < 8; ++r) {
      const int trow = qt * 16 + r + rowoff;
      float a = slo[r] * scale;
      float b = shi[r] * scale;
      if (s0 + col > trow)      a = -__builtin_inff();
      if (s0 + 16 + col > trow) b = -__builtin_inff();
      float cm   = half16_max(fmaxf(a, b));
      float mn   = fmaxf(m[r], cm);
      float resc = __expf(m[r] - mn);
      float plo  = __expf(a - mn);
      float phi  = __expf(b - mn);
      l[r] = l[r] * resc + half16_sum(plo + phi);
      m[r] = mn;
      acc0[r] *= resc; acc1[r] *= resc; acc2[r] *= resc; acc3[r] *= resc;
      const int prow = r + rowoff;
      pbuf[wid][prow * 32 + col]      = (_Float16)plo;   // D-frag -> LDS
      pbuf[wid][prow * 32 + 16 + col] = (_Float16)phi;
    }
    // same-wave LDS is in-order: reload as A-fragment, no barrier needed
    const v16h pa = load_a_frag(&pbuf[wid][0], 0, 0, 32, lane);
    acc0 = wmma_f16(pa, load_b_frag(vh, 0,  s0, SEQ, lane), acc0);
    acc1 = wmma_f16(pa, load_b_frag(vh, 16, s0, SEQ, lane), acc1);
    acc2 = wmma_f16(pa, load_b_frag(vh, 32, s0, SEQ, lane), acc2);
    acc3 = wmma_f16(pa, load_b_frag(vh, 48, s0, SEQ, lane), acc3);
  }

  _Float16* outp = aout + (size_t)bb * SEQ * CH + hn * HD;
#pragma unroll
  for (int r = 0; r < 8; ++r) {
    const int t = qt * 16 + r + rowoff;
    const float inv = 1.0f / l[r];
    _Float16* rowp = outp + (size_t)t * CH;
    rowp[col]      = (_Float16)(acc0[r] * inv);
    rowp[16 + col] = (_Float16)(acc1[r] * inv);
    rowp[32 + col] = (_Float16)(acc2[r] * inv);
    rowp[48 + col] = (_Float16)(acc3[r] * inv);
  }
}

// ---------------- output projection (32x32 tile / wave) ----------------
__global__ __launch_bounds__(256) void gemm_proj_kernel(
    const _Float16* __restrict__ A, const _Float16* __restrict__ Bt,
    const float* __restrict__ bias, float* __restrict__ out) {
  const int lane = threadIdx.x & 31;
  const int gw   = blockIdx.x * 8 + (threadIdx.x >> 5);
  const int m0   = (gw >> 5) * 32;
  const int n0   = (gw & 31) * 32;
  v8f c00 = {}, c01 = {}, c10 = {}, c11 = {};
  for (int k0 = 0; k0 < CH; k0 += 32) {
    __builtin_prefetch(A  + (size_t)(m0 + lane) * CH + k0 + 128, 0, 1);
    __builtin_prefetch(Bt + (size_t)(n0 + lane) * CH + k0 + 128, 0, 1);
    v16h a0 = load_a_frag(A, m0,      k0, CH, lane);
    v16h a1 = load_a_frag(A, m0 + 16, k0, CH, lane);
    v16h b0 = load_b_frag(Bt, n0,      k0, CH, lane);
    v16h b1 = load_b_frag(Bt, n0 + 16, k0, CH, lane);
    c00 = wmma_f16(a0, b0, c00);
    c01 = wmma_f16(a0, b1, c01);
    c10 = wmma_f16(a1, b0, c10);
    c11 = wmma_f16(a1, b1, c11);
  }
  const int rowoff = (lane & 16) ? 8 : 0;
  const int col0 = n0 + (lane & 15), col1 = col0 + 16;
  const float bv0 = bias[col0], bv1 = bias[col1];
#pragma unroll
  for (int r = 0; r < 8; ++r) {
    const int mr0 = m0 + r + rowoff;
    const int mr1 = mr0 + 16;
    out[(size_t)mr0 * CH + col0] = c00[r] + bv0;
    out[(size_t)mr0 * CH + col1] = c01[r] + bv1;
    out[(size_t)mr1 * CH + col0] = c10[r] + bv0;
    out[(size_t)mr1 * CH + col1] = c11[r] + bv1;
  }
}

extern "C" void kernel_launch(void* const* d_in, const int* in_sizes, int n_in,
                              void* d_out, int out_size, void* d_ws,
                              size_t ws_size, hipStream_t stream) {
  (void)in_sizes; (void)n_in; (void)out_size; (void)ws_size;
  const float* x  = (const float*)d_in[0];
  const float* Wq = (const float*)d_in[1];
  const float* bq = (const float*)d_in[2];
  const float* Wk = (const float*)d_in[3];
  const float* bk = (const float*)d_in[4];
  const float* Wv = (const float*)d_in[5];
  const float* bv = (const float*)d_in[6];
  const float* Wp = (const float*)d_in[7];
  const float* bp = (const float*)d_in[8];
  float* out = (float*)d_out;

  char* ws = (char*)d_ws;
  const size_t MiB = 1u << 20;
  _Float16* xh   = (_Float16*)(ws);              // 8 MiB, reused as aout
  _Float16* wqt  = (_Float16*)(ws + 8 * MiB);    // 2 MiB each
  _Float16* wkt  = (_Float16*)(ws + 10 * MiB);
  _Float16* wvt  = (_Float16*)(ws + 12 * MiB);
  _Float16* wpt  = (_Float16*)(ws + 14 * MiB);
  _Float16* qhb  = (_Float16*)(ws + 16 * MiB);   // 8 MiB
  _Float16* khb  = (_Float16*)(ws + 24 * MiB);   // 8 MiB
  _Float16* vthb = (_Float16*)(ws + 32 * MiB);   // 8 MiB
  _Float16* aout = xh;                           // x no longer needed post-QKV

  cvt_f16_kernel<<<1024, 256, 0, stream>>>(x, xh, MROWS * CH);
  cvt_wT_kernel<<<1024, 256, 0, stream>>>(Wq, wqt, CH * CH);
  cvt_wT_kernel<<<1024, 256, 0, stream>>>(Wk, wkt, CH * CH);
  cvt_wT_kernel<<<1024, 256, 0, stream>>>(Wv, wvt, CH * CH);
  cvt_wT_kernel<<<1024, 256, 0, stream>>>(Wp, wpt, CH * CH);

  // 4096 32x32 tiles, 8 waves/block
  gemm_qkv_kernel<<<512, 256, 0, stream>>>(xh, wqt, bq, qhb, 0);
  gemm_qkv_kernel<<<512, 256, 0, stream>>>(xh, wkt, bk, khb, 0);
  gemm_qkv_kernel<<<512, 256, 0, stream>>>(xh, wvt, bv, vthb, 1);

  // 4096 waves = B*N*(T/16)
  attn_kernel<<<512, 256, 0, stream>>>(qhb, khb, vthb, aout);

  gemm_proj_kernel<<<512, 256, 0, stream>>>(aout, wpt, bp, out);
}
